// SpatialGatingUnit_Softmax_12618613915947
// MI455X (gfx1250) — matrix operand order repaired and checked by reading, not compiled, not run on hardware
//
#include <hip/hip_runtime.h>
#include <hip/hip_bf16.h>
#include <math.h>

// ---------------------------------------------------------------------------
// Problem: B=16, S=512, D=2048
//   y = LayerNorm(x);  e = min(b,3)
//   out = GELU_exact(W_e @ y[b] + bias_e) + x        (W_e: 512x512 seq mix)
// Router (Wc, bc) is dead code in the reference -> skipped.
//
// Strategy:
//  - LN kernel writes y TRANSPOSED as bf16: yT[b][d][s]. The 2-byte scattered
//    stores are absorbed by the 192MB L2 (yT = 34MB total), and afterwards
//    BOTH GEMM operands (W row-major MxK, yT row-major NxK) are K-contiguous,
//    matching the WMMA A/B per-lane fragment layout -> no LDS transpose.
//  - GEMM: 128x128 block tile, K-step 32, LDS double-buffered, register-staged
//    b128 global loads overlapping the WMMAs, one barrier per K-step.
// ---------------------------------------------------------------------------

typedef __attribute__((ext_vector_type(16))) __bf16 v16bf;
typedef __attribute__((ext_vector_type(8)))  __bf16 v8bf;
typedef __attribute__((ext_vector_type(8)))  float  v8f;

#define B_DIM 16
#define S_DIM 512
#define D_DIM 2048

__device__ __forceinline__ unsigned short f32_to_bf16_rne(float f) {
    unsigned int u = __float_as_uint(f);
    u += 0x7fffu + ((u >> 16) & 1u);          // round-to-nearest-even
    return (unsigned short)(u >> 16);
}

// ---------------------------------------------------------------------------
// Kernel 1: LayerNorm over D, emit bf16 y TRANSPOSED: yT[b][d][s].
// One 256-thread block per (b,s) row; 8 elements/thread; LDS tree reduction.
// Transposed u16 stores are uncoalesced but land in L2 (34MB << 192MB) and
// write back as full lines.
// ---------------------------------------------------------------------------
__global__ __launch_bounds__(256)
void ln_to_bf16t_kernel(const float* __restrict__ x,
                        const float* __restrict__ gamma,
                        const float* __restrict__ beta,
                        unsigned short* __restrict__ yT) {
    const int row = blockIdx.x;                   // b*S + s
    const int t   = threadIdx.x;
    const float* xr = x + (size_t)row * D_DIM;

    float v[8];
    float s = 0.f, ss = 0.f;
#pragma unroll
    for (int i = 0; i < 8; ++i) {
        float f = xr[t + i * 256];
        v[i] = f; s += f; ss += f * f;
    }

    __shared__ float r1[256], r2[256];
    r1[t] = s; r2[t] = ss;
    __syncthreads();
    for (int off = 128; off > 0; off >>= 1) {
        if (t < off) { r1[t] += r1[t + off]; r2[t] += r2[t + off]; }
        __syncthreads();
    }
    const float mu  = r1[0] * (1.0f / (float)D_DIM);
    const float var = r2[0] * (1.0f / (float)D_DIM) - mu * mu;
    const float inv = rsqrtf(var + 1e-5f);

    const int b = row >> 9;          // row / S_DIM
    const int sidx = row & 511;      // row % S_DIM
    unsigned short* yTb = yT + (size_t)b * D_DIM * S_DIM + sidx;
#pragma unroll
    for (int i = 0; i < 8; ++i) {
        const int d = t + i * 256;
        const float yy = (v[i] - mu) * inv * gamma[d] + beta[d];
        yTb[(size_t)d * S_DIM] = f32_to_bf16_rne(yy);
    }
}

// ---------------------------------------------------------------------------
// Kernel 2: W (4*512*512 fp32) -> bf16, vectorized 4-wide.
// ---------------------------------------------------------------------------
__global__ __launch_bounds__(256)
void w_to_bf16_kernel(const float* __restrict__ W, unsigned short* __restrict__ wb) {
    const int g = blockIdx.x * 256 + threadIdx.x;     // 262144 threads * 4 elems
    const float4 f = ((const float4*)W)[g];
    ushort4 o;
    o.x = f32_to_bf16_rne(f.x);
    o.y = f32_to_bf16_rne(f.y);
    o.z = f32_to_bf16_rne(f.z);
    o.w = f32_to_bf16_rne(f.w);
    ((ushort4*)wb)[g] = o;
}

// ---------------------------------------------------------------------------
// Kernel 3: out[b] = GELU(W_e @ y[b] + bias_e) + x[b]
// Block tile 128(M) x 128(N), K-step 32, LDS double buffered.
// 8 waves as 2(M) x 4(N); each wave: 4x2 WMMA tiles -> 8 wmma / K-step.
//
// Fragment layouts (ISA 7.12.2, wave32, 16-bit):
//   A 16x32: lane (l=lane&15 -> M row, h=lane>>4);
//            elems 0..7  = A[l][h*8+0..7], elems 8..15 = A[l][16+h*8+0..7]
//   B 32x16: lane (n=lane&15 -> N col, h); elems 0..7 = B[h*8+0..7][n], elems 8..15 = B[16+h*8+0..7][n]
//            -> from NxK row-major LDS (yT) this is contiguous too.
//   C/D 16x16 f32: vgpr r -> M = r + 8*h, N = lane&15
// ---------------------------------------------------------------------------
#define LDS_STRIDE 40                 // u16; 80B = 5*16B, keeps 16B alignment
#define BUF_ELEMS  (128 * LDS_STRIDE) // 5120 u16 per buffer

__global__ __launch_bounds__(256)
void moe_seqmix_wmma_kernel(const unsigned short* __restrict__ yTbf,
                            const unsigned short* __restrict__ wbf,
                            const float* __restrict__ x,
                            const float* __restrict__ bias,
                            float* __restrict__ out) {
    const int nt   = blockIdx.x;          // 0..15 : N tile (128 of D)
    const int mt   = blockIdx.y;          // 0..3  : M tile (128 of S)
    const int bidx = blockIdx.z;          // 0..15 : batch
    const int e    = bidx < 3 ? bidx : 3; // expert = min(b,3)

    const int t    = threadIdx.x;
    const int wave = t >> 5;
    const int lane = t & 31;
    const int l16  = lane & 15;
    const int h    = lane >> 4;
    const int wm   = wave & 1;            // 2 waves over M (64 rows each)
    const int wn   = wave >> 1;           // 4 waves over N (32 cols each)

    const int m0 = mt * 128;
    const int n0 = nt * 128;

    // Both operands are row-major with K contiguous; row stride 512 bf16 = 64 uint4.
    const uint4* gA = (const uint4*)(wbf  + (size_t)e    * S_DIM * S_DIM);  // W_e [m][k]
    const uint4* gB = (const uint4*)(yTbf + (size_t)bidx * D_DIM * S_DIM);  // yT  [n][k]

    __shared__ unsigned short ldsA[2 * BUF_ELEMS];   // A[m][k], double buffered
    __shared__ unsigned short ldsB[2 * BUF_ELEMS];   // B^T[n][k], double buffered

    // Staging map: 128 rows x 32 bf16 = 512 uint4 per tile; 2 per thread.
    const int srow = t >> 2;     // 0..63
    const int sc   = t & 3;      // uint4 within 64B row segment

    v8f acc[4][2];
#pragma unroll
    for (int mi = 0; mi < 4; ++mi)
#pragma unroll
        for (int ni = 0; ni < 2; ++ni)
#pragma unroll
            for (int r = 0; r < 8; ++r) acc[mi][ni][r] = 0.0f;

    uint4 ra[2], rb[2];

    // --- prologue: stage K-tile 0 ---
    ra[0] = gA[(size_t)(m0 + srow)      * 64 + sc];
    ra[1] = gA[(size_t)(m0 + srow + 64) * 64 + sc];
    rb[0] = gB[(size_t)(n0 + srow)      * 64 + sc];
    rb[1] = gB[(size_t)(n0 + srow + 64) * 64 + sc];
    *(uint4*)&ldsA[srow        * LDS_STRIDE + sc * 8] = ra[0];
    *(uint4*)&ldsA[(srow + 64) * LDS_STRIDE + sc * 8] = ra[1];
    *(uint4*)&ldsB[srow        * LDS_STRIDE + sc * 8] = rb[0];
    *(uint4*)&ldsB[(srow + 64) * LDS_STRIDE + sc * 8] = rb[1];
    __syncthreads();

    int bufOff = 0;
    for (int it = 0; it < 16; ++it) {
        const int k0n = (it + 1) * 32;

        // Issue next tile's global loads early (overlap with WMMA below).
        if (k0n < S_DIM) {
            const int kc = k0n >> 3;                  // uint4 offset within row
            ra[0] = gA[(size_t)(m0 + srow)      * 64 + kc + sc];
            ra[1] = gA[(size_t)(m0 + srow + 64) * 64 + kc + sc];
            rb[0] = gB[(size_t)(n0 + srow)      * 64 + kc + sc];
            rb[1] = gB[(size_t)(n0 + srow + 64) * 64 + kc + sc];
        }
        if (k0n + 32 < S_DIM) {   // K+2 tile -> L2 (global_prefetch_b8)
            __builtin_prefetch(&gA[(size_t)(m0 + srow) * 64 + ((k0n + 32) >> 3) + sc], 0, 1);
            __builtin_prefetch(&gB[(size_t)(n0 + srow) * 64 + ((k0n + 32) >> 3) + sc], 0, 1);
        }

        // --- compute from current buffer ---
        v16bf af[4], bfr[2];
#pragma unroll
        for (int mi = 0; mi < 4; ++mi) {
            const int m = wm * 64 + mi * 16 + l16;
            const v8bf lo = *(const v8bf*)&ldsA[bufOff + m * LDS_STRIDE + h * 8];
            const v8bf hi = *(const v8bf*)&ldsA[bufOff + m * LDS_STRIDE + 16 + h * 8];
            af[mi] = __builtin_shufflevector(lo, hi, 0, 1, 2, 3, 4, 5, 6, 7,
                                                     8, 9, 10, 11, 12, 13, 14, 15);
        }
#pragma unroll
        for (int ni = 0; ni < 2; ++ni) {
            const int n = wn * 32 + ni * 16 + l16;
            const v8bf lo = *(const v8bf*)&ldsB[bufOff + n * LDS_STRIDE + h * 8];
            const v8bf hi = *(const v8bf*)&ldsB[bufOff + n * LDS_STRIDE + 16 + h * 8];
            bfr[ni] = __builtin_shufflevector(lo, hi, 0, 1, 2, 3, 4, 5, 6, 7,
                                                      8, 9, 10, 11, 12, 13, 14, 15);
        }
#pragma unroll
        for (int mi = 0; mi < 4; ++mi)
#pragma unroll
            for (int ni = 0; ni < 2; ++ni)
                acc[mi][ni] = __builtin_amdgcn_wmma_f32_16x16x32_bf16(
                    false, af[mi], false, bfr[ni], (short)0, acc[mi][ni],
                    false, false);

        // --- store next tile into the other buffer, single barrier ---
        if (k0n < S_DIM) {
            const int ob = bufOff ^ BUF_ELEMS;
            *(uint4*)&ldsA[ob + srow        * LDS_STRIDE + sc * 8] = ra[0];
            *(uint4*)&ldsA[ob + (srow + 64) * LDS_STRIDE + sc * 8] = ra[1];
            *(uint4*)&ldsB[ob + srow        * LDS_STRIDE + sc * 8] = rb[0];
            *(uint4*)&ldsB[ob + (srow + 64) * LDS_STRIDE + sc * 8] = rb[1];
            __syncthreads();
            bufOff = ob;
        }
    }

    // Epilogue: +bias_e[srow], exact GELU, + residual x, fp32 store.
    const float* bp = bias + e * S_DIM;
    const float* xb = x   + (size_t)bidx * S_DIM * D_DIM;
    float*       ob = out + (size_t)bidx * S_DIM * D_DIM;
#pragma unroll
    for (int mi = 0; mi < 4; ++mi) {
#pragma unroll
        for (int ni = 0; ni < 2; ++ni) {
            const int col = n0 + wn * 32 + ni * 16 + l16;
#pragma unroll
            for (int r = 0; r < 8; ++r) {
                const int sr = m0 + wm * 64 + mi * 16 + h * 8 + r;
                const float vsum = acc[mi][ni][r] + bp[sr];
                const float g = 0.5f * vsum *
                                (1.0f + erff(vsum * 0.70710678118654752440f));
                const size_t o = (size_t)sr * D_DIM + col;
                ob[o] = g + xb[o];
            }
        }
    }
}

// ---------------------------------------------------------------------------
// Launch
// ---------------------------------------------------------------------------
extern "C" void kernel_launch(void* const* d_in, const int* in_sizes, int n_in,
                              void* d_out, int out_size, void* d_ws, size_t ws_size,
                              hipStream_t stream) {
    const float* x     = (const float*)d_in[0];   // (16,512,2048) f32
    const float* gamma = (const float*)d_in[1];   // (2048)
    const float* beta  = (const float*)d_in[2];   // (2048)
    // d_in[3] = Wc, d_in[4] = bc : router weights, dead code in reference.
    const float* W     = (const float*)d_in[5];   // (4,512,512)
    const float* bias  = (const float*)d_in[6];   // (4,512)
    float* out = (float*)d_out;                   // (16,512,2048) f32

    // Workspace: yT_bf16 (16*2048*512 u16 = 32MB) then W_bf16 (4*512*512 u16 = 2MB)
    unsigned short* yTbf = (unsigned short*)d_ws;
    unsigned short* wbf  = yTbf + (size_t)B_DIM * D_DIM * S_DIM;

    ln_to_bf16t_kernel<<<dim3(B_DIM * S_DIM), dim3(256), 0, stream>>>(x, gamma, beta, yTbf);
    w_to_bf16_kernel<<<dim3((4 * S_DIM * S_DIM) / (256 * 4)), dim3(256), 0, stream>>>(W, wbf);
    moe_seqmix_wmma_kernel<<<dim3(D_DIM / 128, S_DIM / 128, B_DIM), dim3(256), 0, stream>>>(
        yTbf, wbf, x, bias, out);
}
